// TemporalGRU_74758200754647
// MI455X (gfx1250) — matrix operand order repaired and checked by reading, not compiled, not run on hardware
//
#include <hip/hip_runtime.h>
#include <hip/hip_bf16.h>

// TGCN GRU for MI455X (gfx1250, wave32, WMMA bf16 16x16x32, f32 accumulate)
// Register-tiled: each wave computes a 4(m) x 2(c) block of 16x16 WMMA tiles.
#define NN   1024   // nodes
#define BB   16     // batch
#define TT   24     // time steps
#define FF   2      // input features
#define HH   128    // hidden
#define CATC 160    // (FF+HH)=130 padded to multiple of 32 K-tiles

typedef __attribute__((ext_vector_type(16))) __bf16 v16bf;
typedef __attribute__((ext_vector_type(8)))  float  v8f;

union FragU { uint4 q[2]; v16bf v; };

// float -> bf16 raw bits, round-to-nearest-even (avoids __bf16 scalar conv)
__device__ __forceinline__ unsigned short f2bf(float f) {
  unsigned int u = __float_as_uint(f);
  unsigned int r = (u + 0x7FFFu + ((u >> 16) & 1u)) >> 16;
  return (unsigned short)r;
}

__device__ __forceinline__ float sigmoidf_(float x) { return 1.0f / (1.0f + __expf(-x)); }

// A fragment: 16x32 bf16, row-major A at tile origin `a`, leading dim lda.
// Lanes 0-15 hold M=lane, K {0..7,16..23}; lanes 16-31 hold M=lane-16, K {8..15,24..31}.
__device__ __forceinline__ v16bf load_afrag(const unsigned short* a, int lda, int lane) {
  const int lm = lane & 15, lh = lane >> 4;
  const unsigned short* p = a + (size_t)lm * lda + lh * 8;
  FragU u;
  u.q[0] = *(const uint4*)(p);        // K +0..7
  u.q[1] = *(const uint4*)(p + 16);   // K +16..23
  return u.v;
}

// B fragment: 32x16 bf16 from column-major storage BT[col][k] (ld = ldb).
// Lanes 0-15: col=lane, K {0..7,16..23}; lanes 16-31: col=lane-16, K {8..15,24..31}.
__device__ __forceinline__ v16bf load_bfrag(const unsigned short* bt, int ldb, int lane) {
  const int lc = lane & 15, lh = lane >> 4;
  const unsigned short* p = bt + (size_t)lc * ldb + lh * 8;
  FragU u;
  u.q[0] = *(const uint4*)(p);        // K +0..7
  u.q[1] = *(const uint4*)(p + 16);   // K +16..23
  return u.v;
}

#define WMMA_BF16(A, B, C) \
  __builtin_amdgcn_wmma_f32_16x16x32_bf16(false, (A), false, (B), (short)0, (C), false, false)

// Core: 4x2 tile block. A0 = row-major A at (m0, 0); B0 = col-major B^T at (c0, 0).
// PREFETCH: prefetch A stream (adjacency GEMM only).
template <bool PREFETCH>
__device__ __forceinline__ void mma_4x2(const unsigned short* __restrict__ A0, int lda,
                                        const unsigned short* __restrict__ B0, int ldb,
                                        int K, int lane, v8f acc[4][2]) {
  for (int k = 0; k < K; k += 32) {
    v16bf af[4];
#pragma unroll
    for (int i = 0; i < 4; ++i) af[i] = load_afrag(A0 + (size_t)i * 16 * lda + k, lda, lane);
    v16bf bf0 = load_bfrag(B0 + k, ldb, lane);
    v16bf bf1 = load_bfrag(B0 + (size_t)16 * ldb + k, ldb, lane);
    if (PREFETCH) __builtin_prefetch(A0 + k + 256, 0, 1);  // global_prefetch_b8
#pragma unroll
    for (int i = 0; i < 4; ++i) {
      acc[i][0] = WMMA_BF16(af[i], bf0, acc[i][0]);
      acc[i][1] = WMMA_BF16(af[i], bf1, acc[i][1]);
    }
  }
}

// ---------------- init kernels ----------------

__global__ void cvt_adj_kernel(const float* __restrict__ adj, unsigned short* __restrict__ adjb) {
  int i = blockIdx.x * 256 + threadIdx.x;
  if (i < NN * NN) adjb[i] = f2bf(adj[i]);
}

// wgT(2H x CATC) and wuT(H x CATC): wT[c][k] = w[k][c], zero-padded K rows 130..159
__global__ void prep_w_kernel(const float* __restrict__ wg, const float* __restrict__ wu,
                              unsigned short* __restrict__ wgT, unsigned short* __restrict__ wuT) {
  int i = blockIdx.x * 256 + threadIdx.x;
  const int CIN = FF + HH;  // 130
  if (i < 2 * HH * CATC) {
    int c = i / CATC, k = i % CATC;
    wgT[i] = (k < CIN) ? f2bf(wg[(size_t)k * (2 * HH) + c]) : (unsigned short)0;
  } else if (i < 3 * HH * CATC) {
    int j = i - 2 * HH * CATC;
    int c = j / CATC, k = j % CATC;
    wuT[j] = (k < CIN) ? f2bf(wu[(size_t)k * HH + c]) : (unsigned short)0;
  }
}

__global__ void zero_state_kernel(unsigned short* __restrict__ cat1T,
                                  unsigned short* __restrict__ cat2T,
                                  float* __restrict__ hT) {
  size_t i = (size_t)blockIdx.x * 256 + threadIdx.x;
  const size_t CE = (size_t)BB * CATC * NN;
  if (i < CE) { cat1T[i] = 0; cat2T[i] = 0; }
  if (i < (size_t)BB * HH * NN) hT[i] = 0.0f;
}

// write x_t columns (rows 0,1) of both cat buffers
__global__ void set_x_kernel(const float* __restrict__ x,
                             unsigned short* __restrict__ cat1T,
                             unsigned short* __restrict__ cat2T, int t) {
  int i = blockIdx.x * 256 + threadIdx.x;  // over BB*NN
  if (i >= BB * NN) return;
  int b = i >> 10, n = i & (NN - 1);
  const float* xp = x + (((size_t)b * TT + t) * NN + n) * FF;
  unsigned short x0 = f2bf(xp[0]), x1 = f2bf(xp[1]);
  size_t base = (size_t)b * CATC * NN + n;
  cat1T[base] = x0; cat1T[base + NN] = x1;
  cat2T[base] = x0; cat2T[base + NN] = x1;
}

// ---------------- GEMM kernels (1 wave = 4x2 block of 16x16 D tiles) ----------------

// ax[b] (N x CATC, bf16 row-major) = adj_bf16 (NxN) @ cat[b]  (cat stored col-major catT[b][c][n])
__global__ __launch_bounds__(128)
void gemm_adj_kernel(const unsigned short* __restrict__ adjb,
                     const unsigned short* __restrict__ catT,
                     unsigned short* __restrict__ axout) {
  const int lane = threadIdx.x & 31;
  const int wave = threadIdx.x >> 5;
  const int mg  = blockIdx.x * 4 + wave;  // 0..15 : m-tile group (4 tiles each)
  const int tcg = blockIdx.y;             // 0..4  : c-tile group (2 tiles each)
  const int b   = blockIdx.z;

  const unsigned short* At = adjb + (size_t)mg * 64 * NN;
  const unsigned short* Bt = catT + ((size_t)b * CATC + (size_t)tcg * 32) * NN;

  v8f acc[4][2] = {};
  mma_4x2<true>(At, NN, Bt, NN, NN, lane, acc);

  const int lm = lane & 15, lh = lane >> 4;
#pragma unroll
  for (int i = 0; i < 4; ++i) {
#pragma unroll
    for (int j = 0; j < 2; ++j) {
      unsigned short* D = axout + (size_t)b * NN * CATC +
                          ((size_t)mg * 64 + i * 16 + 8 * lh) * CATC +
                          (tcg * 2 + j) * 16 + lm;
#pragma unroll
      for (int r = 0; r < 8; ++r) D[(size_t)r * CATC] = f2bf(acc[i][j][r]);
    }
  }
}

// gates = sigmoid(ax1[b] @ w_gate + b_gate); epilogue writes r*h -> cat2T (bf16) and u -> uT
__global__ __launch_bounds__(128)
void gemm_gate_kernel(const unsigned short* __restrict__ ax1,
                      const unsigned short* __restrict__ wgT,
                      const float* __restrict__ bgate,
                      const float* __restrict__ hT,
                      unsigned short* __restrict__ cat2T,
                      float* __restrict__ uT) {
  const int lane = threadIdx.x & 31;
  const int wave = threadIdx.x >> 5;
  const int mg  = blockIdx.x * 4 + wave;  // 0..15
  const int tcg = blockIdx.y;             // 0..7 (256 gate cols / 32)
  const int b   = blockIdx.z;

  const unsigned short* At = ax1 + (size_t)b * NN * CATC + (size_t)mg * 64 * CATC;
  const unsigned short* Bt = wgT + (size_t)tcg * 32 * CATC;

  v8f acc[4][2] = {};
  mma_4x2<false>(At, CATC, Bt, CATC, CATC, lane, acc);

  const int lm = lane & 15, lh = lane >> 4;
#pragma unroll
  for (int j = 0; j < 2; ++j) {
    const int cc = tcg * 32 + j * 16 + lm;  // 0..255, uniform side of HH per wave
    const float bias = bgate[cc];
#pragma unroll
    for (int i = 0; i < 4; ++i) {
      const int m0 = mg * 64 + i * 16 + 8 * lh;
      if (cc < HH) {  // reset gate -> r*h into cat2T rows 2..129
        const float* hrow = hT + ((size_t)b * HH + cc) * NN + m0;
        unsigned short* crow = cat2T + ((size_t)b * CATC + 2 + cc) * NN + m0;
#pragma unroll
        for (int r = 0; r < 8; ++r) crow[r] = f2bf(sigmoidf_(acc[i][j][r] + bias) * hrow[r]);
      } else {        // update gate
        float* urow = uT + ((size_t)b * HH + (cc - HH)) * NN + m0;
#pragma unroll
        for (int r = 0; r < 8; ++r) urow[r] = sigmoidf_(acc[i][j][r] + bias);
      }
    }
  }
}

// c = tanh(ax2[b] @ w_update + b); h = u*h + (1-u)*c; write out, hT, next-step cat1T
__global__ __launch_bounds__(128)
void gemm_update_kernel(const unsigned short* __restrict__ ax2,
                        const unsigned short* __restrict__ wuT,
                        const float* __restrict__ bupd,
                        const float* __restrict__ uT,
                        float* __restrict__ hT,
                        unsigned short* __restrict__ cat1T,
                        float* __restrict__ out, int t) {
  const int lane = threadIdx.x & 31;
  const int wave = threadIdx.x >> 5;
  const int mg  = blockIdx.x * 4 + wave;  // 0..15
  const int tcg = blockIdx.y;             // 0..3 (128 cols / 32)
  const int b   = blockIdx.z;

  const unsigned short* At = ax2 + (size_t)b * NN * CATC + (size_t)mg * 64 * CATC;
  const unsigned short* Bt = wuT + (size_t)tcg * 32 * CATC;

  v8f acc[4][2] = {};
  mma_4x2<false>(At, CATC, Bt, CATC, CATC, lane, acc);

  const int lm = lane & 15, lh = lane >> 4;
#pragma unroll
  for (int j = 0; j < 2; ++j) {
    const int cc = tcg * 32 + j * 16 + lm;  // 0..127
    const float bias = bupd[cc];
#pragma unroll
    for (int i = 0; i < 4; ++i) {
      const int m0 = mg * 64 + i * 16 + 8 * lh;
      const float* urow = uT + ((size_t)b * HH + cc) * NN + m0;
      float* hrow = hT + ((size_t)b * HH + cc) * NN + m0;
      unsigned short* crow = cat1T + ((size_t)b * CATC + 2 + cc) * NN + m0;
      float* orow = out + (((size_t)b * TT + t) * NN + m0) * HH + cc;
#pragma unroll
      for (int r = 0; r < 8; ++r) {
        float cv = tanhf(acc[i][j][r] + bias);
        float u  = urow[r];
        float hn = u * hrow[r] + (1.0f - u) * cv;
        hrow[r] = hn;
        crow[r] = f2bf(hn);
        orow[(size_t)r * HH] = hn;
      }
    }
  }
}

// ---------------- host ----------------

extern "C" void kernel_launch(void* const* d_in, const int* in_sizes, int n_in,
                              void* d_out, int out_size, void* d_ws, size_t ws_size,
                              hipStream_t stream) {
  (void)in_sizes; (void)n_in; (void)out_size; (void)ws_size;
  const float* x   = (const float*)d_in[0];
  const float* adj = (const float*)d_in[1];
  const float* wg  = (const float*)d_in[2];
  const float* bg  = (const float*)d_in[3];
  const float* wu  = (const float*)d_in[4];
  const float* bu  = (const float*)d_in[5];
  float* out = (float*)d_out;

  char* ws = (char*)d_ws;
  size_t off = 0;
  auto take = [&](size_t bytes) -> char* {
    char* p = ws + off;
    off = (off + bytes + 255) & ~(size_t)255;
    return p;
  };
  unsigned short* adjb  = (unsigned short*)take((size_t)NN * NN * 2);
  unsigned short* wgT   = (unsigned short*)take((size_t)2 * HH * CATC * 2);
  unsigned short* wuT   = (unsigned short*)take((size_t)HH * CATC * 2);
  unsigned short* cat1T = (unsigned short*)take((size_t)BB * CATC * NN * 2);
  unsigned short* cat2T = (unsigned short*)take((size_t)BB * CATC * NN * 2);
  unsigned short* ax1   = (unsigned short*)take((size_t)BB * NN * CATC * 2);
  unsigned short* ax2   = (unsigned short*)take((size_t)BB * NN * CATC * 2);
  float* hT = (float*)take((size_t)BB * HH * NN * 4);
  float* uT = (float*)take((size_t)BB * HH * NN * 4);

  cvt_adj_kernel<<<(NN * NN + 255) / 256, 256, 0, stream>>>(adj, adjb);
  prep_w_kernel<<<(3 * HH * CATC + 255) / 256, 256, 0, stream>>>(wg, wu, wgT, wuT);
  {
    size_t ztot = (size_t)BB * CATC * NN;
    zero_state_kernel<<<(unsigned)((ztot + 255) / 256), 256, 0, stream>>>(cat1T, cat2T, hT);
  }

  const dim3 gAdj(4, 5, BB);    // 16 m-groups / 4 waves, 5 c-groups, batch
  const dim3 gGate(4, 8, BB);   // 8 c-groups (256 cols)
  const dim3 gUpd(4, 4, BB);    // 4 c-groups (128 cols)

  for (int t = 0; t < TT; ++t) {
    set_x_kernel<<<(BB * NN + 255) / 256, 256, 0, stream>>>(x, cat1T, cat2T, t);
    gemm_adj_kernel<<<gAdj, 128, 0, stream>>>(adjb, cat1T, ax1);
    gemm_gate_kernel<<<gGate, 128, 0, stream>>>(ax1, wgT, bg, hT, cat2T, uT);
    gemm_adj_kernel<<<gAdj, 128, 0, stream>>>(adjb, cat2T, ax2);
    gemm_update_kernel<<<gUpd, 128, 0, stream>>>(ax2, wuT, bu, uT, hT, cat1T, out, t);
  }
}
